// graphNetworkEqvrnt_32091995636059
// MI455X (gfx1250) — matrix operand order repaired and checked by reading, not compiled, not run on hardware
//
#include <hip/hip_runtime.h>
#include <hip/hip_bf16.h>

typedef __attribute__((ext_vector_type(16))) _Float16 v16h;
typedef __attribute__((ext_vector_type(8)))  float    v8f;
typedef __attribute__((ext_vector_type(4)))  unsigned int u32x4;
typedef __attribute__((ext_vector_type(4)))  int      i32x4;
typedef __attribute__((ext_vector_type(8)))  int      i32x8;

#define NNODES 10000
#define NEDGES 160000
#define NCH    192        // xn channels after concat
#define NOP0   64
#define NHID   128
#define NFEAT  385
#define KPAD   416        // 385 padded to 13*32
#define MPAD   400        // 385 padded to 25*16
#define HSTEP  0.1f

// ---------------------------------------------------------------- WMMA frag loaders
// A operand (M x K tile, row-major src, 16x32 f16). ISA layout: lanes 0-15: M=lane,
// halves 0..7 = K 0..7, halves 8..15 = K 16..23 ; lanes 16-31: K 8..15 / 24..31.
__device__ __forceinline__ v16h load_fragA(const _Float16* p, int row_stride, int lane) {
  int r  = lane & 15;
  int k0 = (lane >> 4) << 3;           // 0 or 8
  const _Float16* q = p + (size_t)r * row_stride + k0;
  v16h f;
#pragma unroll
  for (int t = 0; t < 8; ++t) { f[t] = q[t]; f[t + 8] = q[t + 16]; }
  return f;
}
// B operand (K x N tile, stored transposed: rows = N columns, 16x32 f16).
// ISA layout: lanes 0-15: N=lane, halves 0..7 = K 0..7, halves 8..15 = K 16..23 ; lanes 16-31: K 8..15 / 24..31.
__device__ __forceinline__ v16h load_fragB(const _Float16* p, int row_stride, int lane) {
  int r  = lane & 15;
  int k0 = (lane >> 4) << 3;           // 0 or 8
  const _Float16* q = p + (size_t)r * row_stride + k0;
  v16h f;
#pragma unroll
  for (int t = 0; t < 8; ++t) { f[t] = q[t]; f[t + 8] = q[t + 16]; }
  return f;
}
__device__ __forceinline__ v8f zero8() {
  v8f z;
#pragma unroll
  for (int i = 0; i < 8; ++i) z[i] = 0.0f;
  return z;
}

// ---------------------------------------------------------------- TDM 2D tile load
// Packs a Tensor DMA Descriptor (D#) per CDNA5 ISA ch.8 and issues
// tensor_load_to_lds. dims/strides in elements (data_size = 2 bytes).
__device__ __forceinline__ void tdm_load_2d(unsigned lds_off, const void* gptr,
                                            unsigned dim0, unsigned dim1,
                                            unsigned tile0, unsigned tile1,
                                            unsigned stride0) {
  unsigned long long ga = (unsigned long long)(size_t)gptr;
  u32x4 g0;
  g0[0] = 1u;                                              // count=1 (valid user D#)
  g0[1] = lds_off;                                         // lds_addr [63:32]
  g0[2] = (unsigned)(ga & 0xffffffffu);                    // global_addr[31:0]
  g0[3] = (unsigned)((ga >> 32) & 0x1ffffffu) | (2u << 30);// addr[56:32] | type=2
  i32x8 g1;
  g1[0] = (int)(1u << 16);                                 // data_size=2B, mask=0
  g1[1] = (int)((dim0 & 0xffffu) << 16);                   // tensor_dim0[15:0] @63:48
  g1[2] = (int)((dim0 >> 16) | ((dim1 & 0xffffu) << 16));  // dim0[31:16] | dim1[15:0]
  g1[3] = (int)((dim1 >> 16) | (tile0 << 16));             // dim1[31:16] | tile_dim0
  g1[4] = (int)(tile1 & 0xffffu);                          // tile_dim1, tile_dim2=0
  g1[5] = (int)stride0;                                    // tensor_dim0_stride[31:0]
  g1[6] = 0;
  g1[7] = 0;
  i32x4 z4 = {0, 0, 0, 0};
#if defined(__clang_major__) && (__clang_major__ >= 23)
  i32x8 z8 = {0, 0, 0, 0, 0, 0, 0, 0};
  __builtin_amdgcn_tensor_load_to_lds(g0, g1, z4, z4, z8, 0);
#else
  __builtin_amdgcn_tensor_load_to_lds(g0, g1, z4, z4, 0);
#endif
}

// ---------------------------------------------------------------- init coords
__global__ void init_coords_kernel(float* C0, float* C1) {
  int n = blockIdx.x * 256 + threadIdx.x;
  if (n >= NNODES) return;
  for (int c = 0; c < 3; ++c) {
    int v = n - c - 1;
    int q = (v >= 0) ? (v / 3) : -((2 - v) / 3);     // floor div
    int cnt = q + 1; if (cnt < 0) cnt = 0;
    float val = 3.8f * (float)cnt;
    C0[n * 4 + c] = val; C1[n * 4 + c] = val;
  }
  C0[n * 4 + 3] = 0.0f; C1[n * 4 + 3] = 0.0f;
}

// ---------------------------------------------------------------- node open layer
__global__ __launch_bounds__(128) void node_open_kernel(
    const float* __restrict__ xn, const float* __restrict__ K1,
    const float* __restrict__ K2, float* __restrict__ xnw) {
  __shared__ float sK1[NOP0 * 40];
  __shared__ float sK2[NOP0 * NOP0];
  for (int t = threadIdx.x; t < NOP0 * 40;   t += 128) sK1[t] = K1[t];
  for (int t = threadIdx.x; t < NOP0 * NOP0; t += 128) sK2[t] = K2[t];
  __syncthreads();
  int n = blockIdx.x * 128 + threadIdx.x;
  if (n >= NNODES) return;
  float h[NOP0];
#pragma unroll
  for (int c = 0; c < NOP0; ++c) h[c] = 0.0f;
  for (int ci = 0; ci < 40; ++ci) {
    float x = tanhf(xn[(size_t)ci * NNODES + n]);
    for (int co = 0; co < NOP0; ++co) h[co] += sK1[co * 40 + ci] * x;
  }
  float mean = 0.0f;
  for (int c = 0; c < NOP0; ++c) mean += h[c];
  mean *= (1.0f / NOP0);
  float ss = 0.0f;
  for (int c = 0; c < NOP0; ++c) { h[c] -= mean; ss += h[c] * h[c]; }
  float inv = 1.0f / sqrtf(ss + 1e-3f);
  for (int c = 0; c < NOP0; ++c) h[c] = tanhf(h[c] * inv);
  for (int co = 0; co < NOP0; ++co) {
    float y = 0.0f;
    for (int c = 0; c < NOP0; ++c) y += sK2[co * NOP0 + c] * h[c];
    xnw[(size_t)n * NCH + co] = tanhf(y);
  }
}

// ---------------------------------------------------------------- edge open layer + concat scatters
__global__ __launch_bounds__(256) void edge_open_kernel(
    const float* __restrict__ xe, const float* __restrict__ K1E,
    const float* __restrict__ K2E, const int* __restrict__ iInd,
    const int* __restrict__ jInd, float* __restrict__ xe_out,
    float* __restrict__ catdiv, float* __restrict__ catave1,
    float* __restrict__ catave2) {
  __shared__ float sK1[NOP0];
  __shared__ float sK2[NOP0 * NOP0];
  for (int t = threadIdx.x; t < NOP0;        t += 256) sK1[t] = K1E[t];
  for (int t = threadIdx.x; t < NOP0 * NOP0; t += 256) sK2[t] = K2E[t];
  __syncthreads();
  int e = blockIdx.x * 256 + threadIdx.x;
  if (e >= NEDGES) return;
  float t0 = tanhf(xe[e]);
  float h[NOP0];
  float mean = 0.0f;
  for (int c = 0; c < NOP0; ++c) { h[c] = sK1[c] * t0; mean += h[c]; }
  mean *= (1.0f / NOP0);
  float ss = 0.0f;
  for (int c = 0; c < NOP0; ++c) { h[c] -= mean; ss += h[c] * h[c]; }
  float inv = 1.0f / sqrtf(ss + 1e-3f);
  for (int c = 0; c < NOP0; ++c) h[c] = tanhf(h[c] * inv);
  int i = iInd[e], j = jInd[e];
  for (int co = 0; co < NOP0; ++co) {
    float y = 0.0f;
    for (int c = 0; c < NOP0; ++c) y += sK2[co * NOP0 + c] * h[c];
    y = tanhf(y);
    xe_out[(size_t)co * NEDGES + e] = y;
    atomicAdd(&catdiv [i * NOP0 + co],  y);
    atomicAdd(&catdiv [j * NOP0 + co], -y);
    atomicAdd(&catave1[i * NOP0 + co],  y);
    atomicAdd(&catave2[j * NOP0 + co],  y);
  }
}

__global__ void concat_kernel(const float* __restrict__ catdiv,
                              const float* __restrict__ catave1,
                              const float* __restrict__ catave2,
                              float* __restrict__ xnw) {
  int idx = blockIdx.x * 256 + threadIdx.x;        // over NNODES*64
  if (idx >= NNODES * NOP0) return;
  int n = idx / NOP0, c = idx % NOP0;
  xnw[(size_t)n * NCH + 64  + c] = catdiv[idx];
  xnw[(size_t)n * NCH + 128 + c] = fmaxf(catave1[idx], catave2[idx]);
}

// ---------------------------------------------------------------- weight conversions (f32 -> f16, padded)
__global__ void convK_kernel(const float* __restrict__ src, _Float16* __restrict__ dst,
                             int rows_src, int cols_src, int rows_pad, int cols_pad,
                             int total) {
  int idx = blockIdx.x * 256 + threadIdx.x;
  if (idx >= total) return;
  int c = idx % cols_pad;
  int r = (idx / cols_pad) % rows_pad;
  int l = idx / (cols_pad * rows_pad);
  float v = 0.0f;
  if (r < rows_src && c < cols_src)
    v = src[((size_t)l * rows_src + r) * cols_src + c];
  dst[idx] = (_Float16)v;
}

// ---------------------------------------------------------------- Kw1 global sums (once)
__global__ void kwsum_kernel(const float* __restrict__ Kw1, double* sums) {
  __shared__ float r1[256], r2[256];
  int tid = threadIdx.x;
  float v = (tid < NCH) ? Kw1[tid] : 0.0f;
  r1[tid] = v; r2[tid] = v * v;
  __syncthreads();
  for (int s = 128; s > 0; s >>= 1) {
    if (tid < s) { r1[tid] += r1[tid + s]; r2[tid] += r2[tid + s]; }
    __syncthreads();
  }
  if (tid == 0) { sums[4] = (double)r1[0]; sums[5] = (double)r2[0]; }
}

// ---------------------------------------------------------------- edge lengths + partial sums
__global__ __launch_bounds__(256) void edge_len_kernel(
    const float* __restrict__ xnw, const float* __restrict__ Ccur,
    const int* __restrict__ iInd, const int* __restrict__ jInd,
    float* __restrict__ wlen, float* __restrict__ dlen, double* sums) {
  int e = blockIdx.x * 256 + threadIdx.x;
  float wl = 0.0f;
  if (e < NEDGES) {
    int i = iInd[e], j = jInd[e];
    float s = 0.0f;
    for (int c = 0; c < NCH; ++c) {
      float d = xnw[(size_t)i * NCH + c] - xnw[(size_t)j * NCH + c];
      s += d * d;
    }
    wl = sqrtf(s);
    wlen[e] = wl;
    float s2 = 0.0f;
    for (int k = 0; k < 3; ++k) {
      float d = Ccur[i * 4 + k] - Ccur[j * 4 + k];
      s2 += d * d;
    }
    dlen[e] = sqrtf(s2);
  }
  __shared__ float r1[256], r2[256];
  int tid = threadIdx.x;
  r1[tid] = wl; r2[tid] = wl * wl;
  __syncthreads();
  for (int s = 128; s > 0; s >>= 1) {
    if (tid < s) { r1[tid] += r1[tid + s]; r2[tid] += r2[tid + s]; }
    __syncthreads();
  }
  if (tid == 0) { atomicAdd(&sums[0], (double)r1[0]); atomicAdd(&sums[1], (double)r2[0]); }
}

__global__ void wstd_kernel(const double* sums, float* scal) {
  double n  = (double)NCH * (double)NEDGES;
  double S  = sums[4] * sums[0];
  double S2 = sums[5] * sums[1];
  double var = (S2 - S * S / n) / (n - 1.0);
  if (var < 0.0) var = 0.0;
  scal[0] = (float)(1.0 / (sqrt(var) + 1e-4));
}

__global__ void w3std_kernel(const double* sums, float* scal) {
  double n  = 3.0 * (double)NEDGES;
  double var = (sums[3] - sums[2] * sums[2] / n) / (n - 1.0);
  if (var < 0.0) var = 0.0;
  scal[1] = (float)(1.0 / (sqrt(var) + 1e-4));
}

// ---------------------------------------------------------------- build a = tanh(dxe), f16 [E][416]
__global__ __launch_bounds__(256) void build_dxe_kernel(
    const float* __restrict__ xnw, const float* __restrict__ Kw1,
    const float* __restrict__ wlen, const float* __restrict__ dlen,
    const float* __restrict__ scal, const int* __restrict__ iInd,
    const int* __restrict__ jInd, _Float16* __restrict__ A) {
  __shared__ float sKw[NCH];
  for (int t = threadIdx.x; t < NCH; t += 256) sKw[t] = Kw1[t];
  __syncthreads();
  int e = blockIdx.x * 256 + threadIdx.x;
  if (e >= NEDGES) return;
  int i = iInd[e], j = jInd[e];
  float wl = wlen[e] * scal[0];
  _Float16* a = A + (size_t)e * KPAD;
  for (int c = 0; c < NCH; ++c) {
    float xi = xnw[(size_t)i * NCH + c];
    float xj = xnw[(size_t)j * NCH + c];
    float wce = tanhf(sKw[c] * wl);
    a[c]       = (_Float16)tanhf(wce * (xi - xj));
    a[NCH + c] = (_Float16)tanhf(wce * 0.5f * (xi + xj));
  }
  a[384] = (_Float16)tanhf(dlen[e]);
  for (int c = 385; c < KPAD; ++c) a[c] = (_Float16)0.0f;
}

// ---------------------------------------------------------------- GEMM1: H = KE1[128x416] * a  + tv_norm + tanh
// B tile (64 edges x 416) staged into LDS by the Tensor Data Mover.
__global__ __launch_bounds__(256) void gemm1_kernel(
    const _Float16* __restrict__ KE1f,     // [128][416] (this layer)
    const _Float16* __restrict__ Aed,      // [E][416]
    _Float16* __restrict__ Hh) {           // [E][128]
  extern __shared__ _Float16 sB[];         // [64][416]  (53.25 KB dynamic)
  __shared__ float sC[NHID * 65];
  int lane = threadIdx.x & 31;
  int wave = threadIdx.x >> 5;
  size_t e0 = (size_t)blockIdx.x * 64;
  if (threadIdx.x < 32) {                  // wave 0 issues the TDM copy
    tdm_load_2d((unsigned)(size_t)sB, Aed + e0 * KPAD,
                KPAD, 64, KPAD, 64, KPAD);
    __builtin_amdgcn_s_wait_tensorcnt(0);
  }
  __syncthreads();
  v8f acc[4];
#pragma unroll
  for (int n2 = 0; n2 < 4; ++n2) acc[n2] = zero8();
  for (int kk = 0; kk < 13; ++kk) {
    v16h af = load_fragA(KE1f + (size_t)(wave * 16) * KPAD + kk * 32, KPAD, lane);
#pragma unroll
    for (int n2 = 0; n2 < 4; ++n2) {
      v16h bf = load_fragB(sB + (size_t)(n2 * 16) * KPAD + kk * 32, KPAD, lane);
      acc[n2] = __builtin_amdgcn_wmma_f32_16x16x32_f16(
          false, af, false, bf, (short)0, acc[n2], false, false);
    }
  }
  int mbase = wave * 16 + ((lane >> 4) << 3);
  int n = lane & 15;
#pragma unroll
  for (int n2 = 0; n2 < 4; ++n2)
#pragma unroll
    for (int v = 0; v < 8; ++v)
      sC[(mbase + v) * 65 + n2 * 16 + n] = acc[n2][v];
  __syncthreads();
  if (threadIdx.x < 64) {
    int el = threadIdx.x;
    float mean = 0.0f;
    for (int c = 0; c < NHID; ++c) mean += sC[c * 65 + el];
    mean *= (1.0f / NHID);
    float ss = 0.0f;
    for (int c = 0; c < NHID; ++c) { float d = sC[c * 65 + el] - mean; ss += d * d; }
    float inv = 1.0f / sqrtf(ss + 1e-3f);
    _Float16* outp = Hh + (e0 + el) * NHID;
    for (int c = 0; c < NHID; ++c)
      outp[c] = (_Float16)tanhf((sC[c * 65 + el] - mean) * inv);
  }
}

// ---------------------------------------------------------------- GEMM2: dxe = tanh(KE2[400x128] * H), f16 [E][416]
// KE2 (400x128, 102.4 KB) staged into LDS by the Tensor Data Mover.
__global__ __launch_bounds__(256) void gemm2_kernel(
    const _Float16* __restrict__ KE2f,     // [400][128] (this layer)
    const _Float16* __restrict__ Hh,       // [E][128]
    _Float16* __restrict__ Dout) {         // [E][416] (cols 0..399)
  extern __shared__ _Float16 sA2[];        // [400][128] dynamic
  int lane = threadIdx.x & 31;
  int wave = threadIdx.x >> 5;
  size_t e0 = (size_t)blockIdx.x * 128 + wave * 16;
  if (threadIdx.x < 32) {                  // wave 0 issues the TDM copy
    tdm_load_2d((unsigned)(size_t)sA2, KE2f, NHID, MPAD, NHID, MPAD, NHID);
    __builtin_amdgcn_s_wait_tensorcnt(0);
  }
  v16h bf[4];
#pragma unroll
  for (int kk = 0; kk < 4; ++kk)
    bf[kk] = load_fragB(Hh + e0 * NHID + kk * 32, NHID, lane);
  __syncthreads();
  int voff = (lane >> 4) << 3;
  int n = lane & 15;
  for (int m2 = 0; m2 < 25; ++m2) {
    v8f acc = zero8();
#pragma unroll
    for (int kk = 0; kk < 4; ++kk) {
      v16h af = load_fragA(sA2 + (size_t)(m2 * 16) * NHID + kk * 32, NHID, lane);
      acc = __builtin_amdgcn_wmma_f32_16x16x32_f16(
          false, af, false, bf[kk], (short)0, acc, false, false);
    }
    _Float16* outp = Dout + (e0 + n) * KPAD + m2 * 16 + voff;
#pragma unroll
    for (int v = 0; v < 8; ++v) outp[v] = (_Float16)tanhf(acc[v]);
  }
}

// ---------------------------------------------------------------- w3 = Kw2 * dxe + partial sums
__global__ __launch_bounds__(256) void w3_kernel(
    const _Float16* __restrict__ D, const float* __restrict__ Kw2,
    float* __restrict__ w3raw, double* sums) {
  __shared__ float sKw2[3 * NFEAT];
  for (int t = threadIdx.x; t < 3 * NFEAT; t += 256) sKw2[t] = Kw2[t];
  __syncthreads();
  int e = blockIdx.x * 256 + threadIdx.x;
  float ps = 0.0f, ps2 = 0.0f;
  if (e < NEDGES) {
    const _Float16* d = D + (size_t)e * KPAD;
    float a0 = 0.0f, a1 = 0.0f, a2 = 0.0f;
    for (int c = 0; c < NFEAT; ++c) {
      float x = (float)d[c];
      a0 += sKw2[c] * x;
      a1 += sKw2[NFEAT + c] * x;
      a2 += sKw2[2 * NFEAT + c] * x;
    }
    w3raw[e * 3 + 0] = a0; w3raw[e * 3 + 1] = a1; w3raw[e * 3 + 2] = a2;
    ps  = a0 + a1 + a2;
    ps2 = a0 * a0 + a1 * a1 + a2 * a2;
  }
  __shared__ float r1[256], r2[256];
  int tid = threadIdx.x;
  r1[tid] = ps; r2[tid] = ps2;
  __syncthreads();
  for (int s = 128; s > 0; s >>= 1) {
    if (tid < s) { r1[tid] += r1[tid + s]; r2[tid] += r2[tid + s]; }
    __syncthreads();
  }
  if (tid == 0) { atomicAdd(&sums[2], (double)r1[0]); atomicAdd(&sums[3], (double)r2[0]); }
}

// ---------------------------------------------------------------- coords edge scatter
__global__ void coords_kernel(const float* __restrict__ w3raw,
                              const float* __restrict__ scal,
                              const float* __restrict__ Ccur,
                              const int* __restrict__ iInd,
                              const int* __restrict__ jInd,
                              float* __restrict__ cave1, float* __restrict__ cave2) {
  int e = blockIdx.x * 256 + threadIdx.x;
  if (e >= NEDGES) return;
  int i = iInd[e], j = jInd[e];
  float s3 = scal[1];
  for (int k = 0; k < 3; ++k) {
    float w3 = tanhf(w3raw[e * 3 + k] * s3);
    float g  = w3 * (Ccur[i * 4 + k] - Ccur[j * 4 + k]);
    float v  = w3 * g;
    atomicAdd(&cave1[i * 4 + k], v);
    atomicAdd(&cave2[j * 4 + k], v);
  }
}

// ---------------------------------------------------------------- xn edge scatter (divE / aveE)
__global__ __launch_bounds__(256) void scatterE_kernel(
    const _Float16* __restrict__ D, const float* __restrict__ Kw1,
    const float* __restrict__ wlen, const float* __restrict__ scal,
    const int* __restrict__ iInd, const int* __restrict__ jInd,
    float* __restrict__ ndiv, float* __restrict__ nave1, float* __restrict__ nave2) {
  __shared__ float sKw[NCH];
  for (int t = threadIdx.x; t < NCH; t += 256) sKw[t] = Kw1[t];
  __syncthreads();
  int e = blockIdx.x * 256 + threadIdx.x;
  if (e >= NEDGES) return;
  int i = iInd[e], j = jInd[e];
  float wl = wlen[e] * scal[0];
  const _Float16* d = D + (size_t)e * KPAD;
  for (int c = 0; c < NCH; ++c) {
    float wce = tanhf(sKw[c] * wl);
    float vd = wce * (float)d[c];
    atomicAdd(&ndiv[(size_t)i * NCH + c],  vd);
    atomicAdd(&ndiv[(size_t)j * NCH + c], -vd);
    float va = wce * (float)d[NCH + c];
    atomicAdd(&nave1[(size_t)i * NCH + c], va);
    atomicAdd(&nave2[(size_t)j * NCH + c], va);
  }
}

// ---------------------------------------------------------------- Verlet + xn update
__global__ void update_kernel(float* __restrict__ Cold,
                              const float* __restrict__ cave1,
                              const float* __restrict__ cave2,
                              float* __restrict__ xnw,
                              const float* __restrict__ ndiv,
                              const float* __restrict__ nave1,
                              const float* __restrict__ nave2) {
  int n = blockIdx.x * 256 + threadIdx.x;
  if (n >= NNODES) return;
  for (int k = 0; k < 3; ++k)
    Cold[n * 4 + k] += 2.0f * HSTEP * fmaxf(cave1[n * 4 + k], cave2[n * 4 + k]);
  for (int c = 0; c < NCH; ++c)
    xnw[(size_t)n * NCH + c] -=
        HSTEP * (ndiv[(size_t)n * NCH + c] +
                 fmaxf(nave1[(size_t)n * NCH + c], nave2[(size_t)n * NCH + c]));
}

// ---------------------------------------------------------------- final transpose to d_out
__global__ void final_kernel(const float* __restrict__ Ccur,
                             const float* __restrict__ xnw,
                             float* __restrict__ out) {
  int n = blockIdx.x * 256 + threadIdx.x;
  if (n >= NNODES) return;
  for (int k = 0; k < 3; ++k) out[(size_t)k * NNODES + n] = Ccur[n * 4 + k];
  for (int c = 0; c < NCH; ++c)
    out[3 * NNODES + (size_t)c * NNODES + n] = xnw[(size_t)n * NCH + c];
}

// ================================================================ host
extern "C" void kernel_launch(void* const* d_in, const int* in_sizes, int n_in,
                              void* d_out, int out_size, void* d_ws, size_t ws_size,
                              hipStream_t stream) {
  const float* xn_in = (const float*)d_in[0];
  const float* xe_in = (const float*)d_in[1];
  const float* K1N   = (const float*)d_in[2];
  const float* K2N   = (const float*)d_in[3];
  const float* K1E   = (const float*)d_in[4];
  const float* K2E   = (const float*)d_in[5];
  const float* KE1   = (const float*)d_in[6];
  const float* KE2   = (const float*)d_in[7];
  const float* Kw1   = (const float*)d_in[8];
  const float* Kw2   = (const float*)d_in[9];
  const int*   iInd  = (const int*)d_in[10];
  const int*   jInd  = (const int*)d_in[11];
  float* out = (float*)d_out;
  char* ws = (char*)d_ws;

  // workspace layout (all sizes 256B multiples except sums: 64 B)
  size_t off = 0;
  auto take = [&](size_t bytes) { size_t o = off; off += bytes; return o; };
  float*    xnw     = (float*)(ws + take((size_t)NNODES * NCH * 4));
  float*    catdiv  = (float*)(ws + take((size_t)NNODES * NOP0 * 4));
  float*    catave1 = (float*)(ws + take((size_t)NNODES * NOP0 * 4));
  float*    catave2 = (float*)(ws + take((size_t)NNODES * NOP0 * 4));
  float*    C0      = (float*)(ws + take((size_t)NNODES * 4 * 4));
  float*    C1      = (float*)(ws + take((size_t)NNODES * 4 * 4));
  float*    cave1   = (float*)(ws + take((size_t)NNODES * 4 * 4));
  float*    cave2   = (float*)(ws + take((size_t)NNODES * 4 * 4));
  float*    ndiv    = (float*)(ws + take((size_t)NNODES * NCH * 4));
  float*    nave1   = (float*)(ws + take((size_t)NNODES * NCH * 4));
  float*    nave2   = (float*)(ws + take((size_t)NNODES * NCH * 4));
  float*    wlen    = (float*)(ws + take((size_t)NEDGES * 4));
  float*    dlen    = (float*)(ws + take((size_t)NEDGES * 4));
  float*    w3raw   = (float*)(ws + take((size_t)NEDGES * 3 * 4));
  _Float16* KE1f    = (_Float16*)(ws + take((size_t)4 * NHID * KPAD * 2));
  _Float16* KE2f    = (_Float16*)(ws + take((size_t)4 * MPAD * NHID * 2));
  _Float16* AD      = (_Float16*)(ws + take((size_t)NEDGES * KPAD * 2)); // a / dxe alias
  _Float16* Hh      = (_Float16*)(ws + take((size_t)NEDGES * NHID * 2));
  double*   sums    = (double*)(ws + take(8 * sizeof(double)));
  float*    scal    = (float*)(ws + take(256));

  const int EB = (NEDGES + 255) / 256;  // 625
  const int NB = (NNODES + 255) / 256;  // 40

  // ---- open layers + concat
  hipMemsetAsync(catdiv, 0, (size_t)3 * NNODES * NOP0 * 4, stream);
  init_coords_kernel<<<NB, 256, 0, stream>>>(C0, C1);
  node_open_kernel<<<(NNODES + 127) / 128, 128, 0, stream>>>(xn_in, K1N, K2N, xnw);
  edge_open_kernel<<<EB, 256, 0, stream>>>(xe_in, K1E, K2E, iInd, jInd,
                                           out + 3 * NNODES + (size_t)NCH * NNODES,
                                           catdiv, catave1, catave2);
  concat_kernel<<<(NNODES * NOP0 + 255) / 256, 256, 0, stream>>>(catdiv, catave1,
                                                                 catave2, xnw);
  kwsum_kernel<<<1, 256, 0, stream>>>(Kw1, sums);
  {
    int tot1 = 4 * NHID * KPAD;
    convK_kernel<<<(tot1 + 255) / 256, 256, 0, stream>>>(KE1, KE1f, NHID, NFEAT,
                                                         NHID, KPAD, tot1);
    int tot2 = 4 * MPAD * NHID;
    convK_kernel<<<(tot2 + 255) / 256, 256, 0, stream>>>(KE2, KE2f, NFEAT, NHID,
                                                         MPAD, NHID, tot2);
  }

  // ---- layer loop
  float* Ccur = C0;
  float* Cold = C1;
  for (int l = 0; l < 4; ++l) {
    hipMemsetAsync(sums, 0, 4 * sizeof(double), stream);
    edge_len_kernel<<<EB, 256, 0, stream>>>(xnw, Ccur, iInd, jInd, wlen, dlen, sums);
    wstd_kernel<<<1, 1, 0, stream>>>(sums, scal);
    build_dxe_kernel<<<EB, 256, 0, stream>>>(xnw, Kw1, wlen, dlen, scal, iInd, jInd, AD);
    gemm1_kernel<<<NEDGES / 64, 256, (size_t)64 * KPAD * 2, stream>>>(
        KE1f + (size_t)l * NHID * KPAD, AD, Hh);
    gemm2_kernel<<<NEDGES / 128, 256, (size_t)MPAD * NHID * 2, stream>>>(
        KE2f + (size_t)l * MPAD * NHID, Hh, AD);
    w3_kernel<<<EB, 256, 0, stream>>>(AD, Kw2, w3raw, sums);
    w3std_kernel<<<1, 1, 0, stream>>>(sums, scal);
    hipMemsetAsync(cave1, 0, (size_t)2 * NNODES * 4 * 4, stream);
    coords_kernel<<<EB, 256, 0, stream>>>(w3raw, scal, Ccur, iInd, jInd, cave1, cave2);
    hipMemsetAsync(ndiv, 0, (size_t)3 * NNODES * NCH * 4, stream);
    scatterE_kernel<<<EB, 256, 0, stream>>>(AD, Kw1, wlen, scal, iInd, jInd,
                                            ndiv, nave1, nave2);
    update_kernel<<<NB, 256, 0, stream>>>(Cold, cave1, cave2, xnw, ndiv, nave1, nave2);
    float* t = Ccur; Ccur = Cold; Cold = t;
  }

  final_kernel<<<NB, 256, 0, stream>>>(Ccur, xnw, out);
  (void)in_sizes; (void)n_in; (void)out_size; (void)ws_size;
}
